// RN_19344532701281
// MI455X (gfx1250) — compile-verified, matrix operand order checked by reading, not run on hardware
//
#include <hip/hip_runtime.h>

// Relation Network forward for MI455X (gfx1250, wave32, WMMA).
// Dominant GEMM (131072 x 511 x 512 after factorization) runs on
// v_wmma_f32_16x16x32_bf16 with register-resident B fragments and
// LDS-staged, on-the-fly-generated A tiles (h = relu(u + v)).

#define B_   8
#define D_   128
#define K_   7
#define Q_   256
#define G0_  511
#define G0P  512   // padded K for WMMA
#define G1_  512
#define F1_  512
#define F2_  1024
#define OUT_ 16

#define MCHUNKS 64
#define TILES_PER_CHUNK 8   // (D_*D_/32) / MCHUNKS = 512/64

typedef __attribute__((ext_vector_type(16))) __bf16 v16bf;
typedef __attribute__((ext_vector_type(8)))  float  v8f;

__device__ __forceinline__ unsigned short f2bf(float f) {
  union { float f; unsigned int u; } x; x.f = f;
  unsigned int u = x.u;
  unsigned int r = u + 0x7FFFu + ((u >> 16) & 1u);   // round-to-nearest-even
  return (unsigned short)(r >> 16);
}

// ---- prep: u[b,d,n] = x[b,d,:] @ g0_w[0:7, n] + g0_b[n]
//           v[b,d,n] = x[b,d,:] @ g0_w[7:14, n]        (pad col 511 -> 0)
__global__ void uv_kernel(const float* __restrict__ x,
                          const float* __restrict__ g0w,
                          const float* __restrict__ g0b,
                          float* __restrict__ u, float* __restrict__ v) {
  int idx = blockIdx.x * blockDim.x + threadIdx.x;   // B*D*G0P
  if (idx >= B_ * D_ * G0P) return;
  int n  = idx & (G0P - 1);
  int bd = idx >> 9;
  float su = 0.f, sv = 0.f;
  if (n < G0_) {
    const float* xp = x + bd * K_;
#pragma unroll
    for (int k = 0; k < K_; ++k) {
      float xv = xp[k];
      su += xv * g0w[k * G0_ + n];
      sv += xv * g0w[(K_ + k) * G0_ + n];
    }
    su += g0b[n];
  }
  u[idx] = (n < G0_) ? su : 0.f;
  v[idx] = (n < G0_) ? sv : 0.f;
}

// ---- prep: qc[b,n] = qst[b,:] @ g1_w[511:767, n] + g1_b[n]
__global__ void qc_kernel(const float* __restrict__ qst,
                          const float* __restrict__ g1w,
                          const float* __restrict__ g1b,
                          float* __restrict__ qc) {
  int idx = blockIdx.x * blockDim.x + threadIdx.x;   // B*G1
  if (idx >= B_ * G1_) return;
  int n = idx & (G1_ - 1);
  int b = idx >> 9;
  float s = g1b[n];
  const float* q = qst + b * Q_;
  for (int j = 0; j < Q_; ++j)
    s += q[j] * g1w[(G0_ + j) * G1_ + n];
  qc[idx] = s;
}

// ---- prep: pack g1_w[0:511, :] -> bf16 WMMA B fragments, fragment-major.
// 16-bit B 32x16 lane layout: lane = half*16 + (n%16); VGPR v holds
// K = 2v, 2v+1 (v<4) or 2v+8, 2v+9 (v>=4), +8 for lanes 16-31.  Fragment index = (strip*16+kt)*32+lane.
__global__ void packw_kernel(const float* __restrict__ g1w,
                             unsigned short* __restrict__ wf) {
  int idx = blockIdx.x * blockDim.x + threadIdx.x;   // 32*16*32*16
  if (idx >= 32 * 16 * 32 * 16) return;
  int e     = idx & 15;
  int lane  = (idx >> 4) & 31;
  int kt    = (idx >> 9) & 15;
  int strip = idx >> 13;
  int vv = e >> 1, pos = e & 1;
  int kof = (vv < 4) ? (2 * vv) : (16 + 2 * (vv - 4));
  int kg = kt * 32 + (lane >> 4) * 8 + kof + pos;    // symmetric A/B K-mapping
  int n  = strip * 16 + (lane & 15);
  float val = (kg < G0_) ? g1w[kg * G1_ + n] : 0.f;
  wf[idx] = f2bf(val);
}

__global__ void zero_kernel(float* __restrict__ p, int n) {
  int i = blockIdx.x * blockDim.x + threadIdx.x;
  if (i < n) p[i] = 0.f;
}

// ---- main: x_g[b,n] = sum over 16384 pairs of relu( h_pair @ W1h + qc[b,n] )
// grid = (MCHUNKS, 4 n-blocks, B); block = 256 (8 waves, one 16-col strip each)
__global__ void __launch_bounds__(256)
rn_g1_kernel(const float* __restrict__ u, const float* __restrict__ v,
             const float* __restrict__ qc, const unsigned short* __restrict__ wf,
             float* __restrict__ xg) {
  __shared__ unsigned short hs[32][520];   // 32 pair-rows x 512 K (stride 520: bank-safe)

  const int b     = blockIdx.z;
  const int chunk = blockIdx.x;
  const int tid   = threadIdx.x;
  const int wave  = tid >> 5;
  const int lane  = tid & 31;
  const int strip = blockIdx.y * 8 + wave;          // 0..31 -> 16 columns each

  // Register-resident B fragments for this wave's strip (K = 0..511).
  const v16bf* wp = ((const v16bf*)wf) + (strip * 16) * 32 + lane;
  v16bf Bf[16];
#pragma unroll
  for (int kt = 0; kt < 16; ++kt) Bf[kt] = wp[kt * 32];

  const int nglob = strip * 16 + (lane & 15);
  const float qcv = qc[b * G1_ + nglob];
  const float* ub = u + b * D_ * G0P;
  const float* vb = v + b * D_ * G0P;

  float s = 0.f;
  const int m0   = lane & 15;
  const int half = lane >> 4;

  for (int t = 0; t < TILES_PER_CHUNK; ++t) {
    const int mbase = (chunk * TILES_PER_CHUNK + t) * 32;
    __syncthreads();
    // Cooperative fill: h[row,k..k+7] = relu(u[b,c,k..] + v[b,a,k..]) as bf16,
    // native (__bf16) converts + one ds_store_b128 per 8 elements.
#pragma unroll
    for (int it = 0; it < 8; ++it) {
      int chunk8 = tid + it * 256;         // 2048 chunks of 8 halves
      int row = chunk8 >> 6;               // 64 chunks per 512-half row
      int k   = (chunk8 & 63) * 8;
      int p   = mbase + row;
      int a = p >> 7, c = p & 127;
      const float4* ur = (const float4*)(ub + c * G0P + k);
      const float4* vr = (const float4*)(vb + a * G0P + k);
      float4 u0 = ur[0], u1 = ur[1];
      float4 v0 = vr[0], v1 = vr[1];
      union { __bf16 h[8]; uint4 q; } pk;
      pk.h[0] = (__bf16)fmaxf(u0.x + v0.x, 0.f);
      pk.h[1] = (__bf16)fmaxf(u0.y + v0.y, 0.f);
      pk.h[2] = (__bf16)fmaxf(u0.z + v0.z, 0.f);
      pk.h[3] = (__bf16)fmaxf(u0.w + v0.w, 0.f);
      pk.h[4] = (__bf16)fmaxf(u1.x + v1.x, 0.f);
      pk.h[5] = (__bf16)fmaxf(u1.y + v1.y, 0.f);
      pk.h[6] = (__bf16)fmaxf(u1.z + v1.z, 0.f);
      pk.h[7] = (__bf16)fmaxf(u1.w + v1.w, 0.f);
      *(uint4*)&hs[row][k] = pk.q;         // 16B-aligned: 1040*row + 2*k
    }
    __syncthreads();

    v8f acc0 = {}; v8f acc1 = {};
    const unsigned short* r0 = &hs[m0][0];
    const unsigned short* r1 = &hs[m0 + 16][0];
#pragma unroll
    for (int kt = 0; kt < 16; ++kt) {
      const int k0 = kt * 32 + half * 8;
      union { v16bf bv; uint4 q[2]; } A0, A1;   // 16-bit A lane map (ISA 7.12.2)
      A0.q[0] = *(const uint4*)(r0 + k0);
      A0.q[1] = *(const uint4*)(r0 + k0 + 16);
      A1.q[0] = *(const uint4*)(r1 + k0);
      A1.q[1] = *(const uint4*)(r1 + k0 + 16);
      acc0 = __builtin_amdgcn_wmma_f32_16x16x32_bf16(false, A0.bv, false, Bf[kt],
                                                     (short)0, acc0, false, false);
      acc1 = __builtin_amdgcn_wmma_f32_16x16x32_bf16(false, A1.bv, false, Bf[kt],
                                                     (short)0, acc1, false, false);
    }
    // +qc, ReLU, reduce this tile's 16 M-rows per lane in-register.
#pragma unroll
    for (int i = 0; i < 8; ++i) {
      float a0 = acc0[i] + qcv; s += a0 > 0.f ? a0 : 0.f;
      float a1 = acc1[i] + qcv; s += a1 > 0.f ? a1 : 0.f;
    }
  }
  atomicAdd(&xg[b * G1_ + nglob], s);
}

// ---- tail: f1 -> relu -> f2 -> relu -> f3 -> log_softmax (tiny)
__global__ void __launch_bounds__(256)
tail_kernel(const float* __restrict__ xg,
            const float* __restrict__ f1w, const float* __restrict__ f1b,
            const float* __restrict__ f2w, const float* __restrict__ f2b,
            const float* __restrict__ f3w, const float* __restrict__ f3b,
            float* __restrict__ out) {
  __shared__ float t0[F1_];
  __shared__ float t1[F2_];
  __shared__ float lg[OUT_];
  __shared__ float red[2];
  const int b = blockIdx.x, tid = threadIdx.x;
  const float* xb = xg + b * G1_;
  for (int n = tid; n < F1_; n += 256) {
    float a = f1b[n];
    for (int k = 0; k < G1_; ++k) a += xb[k] * f1w[k * F1_ + n];
    t0[n] = a > 0.f ? a : 0.f;
  }
  __syncthreads();
  for (int n = tid; n < F2_; n += 256) {
    float a = f2b[n];
    for (int k = 0; k < F1_; ++k) a += t0[k] * f2w[k * F2_ + n];
    t1[n] = a > 0.f ? a : 0.f;
  }
  __syncthreads();
  if (tid < OUT_) {
    float a = f3b[tid];
    for (int k = 0; k < F2_; ++k) a += t1[k] * f3w[k * OUT_ + tid];
    lg[tid] = a;
  }
  __syncthreads();
  if (tid == 0) {
    float mx = lg[0];
    for (int i = 1; i < OUT_; ++i) mx = lg[i] > mx ? lg[i] : mx;
    float se = 0.f;
    for (int i = 0; i < OUT_; ++i) se += __expf(lg[i] - mx);
    red[0] = mx; red[1] = __logf(se);
  }
  __syncthreads();
  if (tid < OUT_) out[b * OUT_ + tid] = lg[tid] - red[0] - red[1];
}

extern "C" void kernel_launch(void* const* d_in, const int* in_sizes, int n_in,
                              void* d_out, int out_size, void* d_ws, size_t ws_size,
                              hipStream_t stream) {
  (void)in_sizes; (void)n_in; (void)out_size; (void)ws_size;
  const float* x   = (const float*)d_in[0];
  const float* qst = (const float*)d_in[1];
  const float* g0w = (const float*)d_in[2];
  const float* g0b = (const float*)d_in[3];
  const float* g1w = (const float*)d_in[4];
  const float* g1b = (const float*)d_in[5];
  const float* f1w = (const float*)d_in[6];
  const float* f1b = (const float*)d_in[7];
  const float* f2w = (const float*)d_in[8];
  const float* f2b = (const float*)d_in[9];
  const float* f3w = (const float*)d_in[10];
  const float* f3b = (const float*)d_in[11];
  float* out = (float*)d_out;

  char* ws = (char*)d_ws;
  unsigned short* wfrag = (unsigned short*)ws; ws += (size_t)32*16*32*16*sizeof(unsigned short);
  float* u  = (float*)ws; ws += (size_t)B_*D_*G0P*sizeof(float);
  float* v  = (float*)ws; ws += (size_t)B_*D_*G0P*sizeof(float);
  float* qc = (float*)ws; ws += (size_t)B_*G1_*sizeof(float);
  float* xg = (float*)ws; ws += (size_t)B_*G1_*sizeof(float);

  uv_kernel<<<(B_*D_*G0P + 255) / 256, 256, 0, stream>>>(x, g0w, g0b, u, v);
  qc_kernel<<<(B_*G1_ + 255) / 256, 256, 0, stream>>>(qst, g1w, g1b, qc);
  packw_kernel<<<(32*16*32*16 + 255) / 256, 256, 0, stream>>>(g1w, wfrag);
  zero_kernel<<<(B_*G1_ + 255) / 256, 256, 0, stream>>>(xg, B_*G1_);

  dim3 grid(MCHUNKS, 4, B_);
  rn_g1_kernel<<<grid, 256, 0, stream>>>(u, v, qc, wfrag, xg);

  tail_kernel<<<B_, 256, 0, stream>>>(xg, f1w, f1b, f2w, f2b, f3w, f3b, out);
}
